// MHCtxAttention_19207093748543
// MI455X (gfx1250) — matrix operand order repaired and checked by reading, compiled, not run on hardware
//
#include <hip/hip_runtime.h>
#include <stdint.h>
#include <stddef.h>

// Problem dims (fixed by reference)
#define N_      16
#define T_      2000
#define ED_     512    // ENC_DIM == DEC_DIM == ATT_DIM
#define H_      4
#define A_      512
#define INNER_  2048
#define TSEG_   8
#define MT_     32     // t-rows per block in the score kernel

typedef __bf16 v8bf  __attribute__((ext_vector_type(8)));
typedef __bf16 v16bf __attribute__((ext_vector_type(16)));
typedef float  v8f   __attribute__((ext_vector_type(8)));
typedef int    v4i   __attribute__((ext_vector_type(4)));

typedef __attribute__((address_space(1))) v4i* as1_v4i_ptr;
typedef __attribute__((address_space(3))) v4i* as3_v4i_ptr;

#if defined(__AMDGCN__) && __has_builtin(__builtin_amdgcn_global_load_async_to_lds_b128)
#define HAVE_ASYNC_LDS 1
#else
#define HAVE_ASYNC_LDS 0
#endif

static __device__ __forceinline__ uint16_t f32_to_bf16_rne(float f) {
  union { float f; uint32_t u; } c; c.f = f;
  uint32_t u = c.u;
  uint32_t r = u + 0x7FFFu + ((u >> 16) & 1u);
  return (uint16_t)(r >> 16);
}

// ---------------- fp32 -> bf16 conversion (x4 vectorized) ----------------
__global__ void cvt_bf16_kernel(const float* __restrict__ src,
                                uint16_t* __restrict__ dst, int n4) {
  int i = blockIdx.x * blockDim.x + threadIdx.x;  // in float4 units
  if (i < n4) {
    float4 v = ((const float4*)src)[i];
    ushort4 o;
    o.x = f32_to_bf16_rne(v.x);
    o.y = f32_to_bf16_rne(v.y);
    o.z = f32_to_bf16_rne(v.z);
    o.w = f32_to_bf16_rne(v.w);
    ((ushort4*)dst)[i] = o;
  }
}

// ---------------- dec_part = dec_prev @ W_dec^T  [N, INNER] ----------------
__global__ void decpart_kernel(const float* __restrict__ dec_prev,
                               const float* __restrict__ W_dec,
                               float* __restrict__ dp) {
  int idx = blockIdx.x * blockDim.x + threadIdx.x;  // N*INNER = 32768
  int n = idx >> 11;
  int e = idx & (INNER_ - 1);
  const float* x = dec_prev + (size_t)n * ED_;
  const float* w = W_dec + (size_t)e * ED_;
  float s = 0.f;
#pragma unroll 8
  for (int d = 0; d < ED_; ++d) s += x[d] * w[d];
  dp[idx] = s;
}

// ---------------- fused kp-GEMM + tanh + per-head contraction -> score ----
// grid (ceil(T/32), N, H), block 256 (8 waves). Each wave owns 64 columns of
// head h and BOTH 16-row tiles of the block's 32 t-rows (B reuse x2).
// The 32x512 bf16 A tile is staged in LDS once per block (async if available).
__global__ __launch_bounds__(256) void score_kernel(
    const uint16_t* __restrict__ encb,   // [N,T,ED] bf16
    const uint16_t* __restrict__ wkeyb,  // [INNER,ED] bf16
    const float* __restrict__ dp,        // [N,INNER]
    const float* __restrict__ w_conv,    // [H,A]
    float* __restrict__ score)           // [N,H,T]
{
  const int t0   = blockIdx.x * MT_;
  const int n    = blockIdx.y;
  const int h    = blockIdx.z;
  const int tid  = threadIdx.x;
  const int wave = tid >> 5;
  const int lane = tid & 31;
  const int lh   = lane & 15;   // row (A) / column (B,C) within 16-tile
  const int hb   = lane >> 4;   // K-half select

  __shared__ v4i   atile_i4[MT_ * ED_ / 8];   // 32 KB: [32 rows][512 bf16]
  __shared__ float part[8][MT_];

  // ---- stage A tile: rows t0..t0+31 (clamped to T-1 for the tail block) ----
  {
    const v4i* gsrc = (const v4i*)encb;       // 64 v4i per row
    for (int i = tid; i < MT_ * ED_ / 8; i += 256) {
      int row  = i >> 6;
      int cir  = i & 63;
      int trow = t0 + row;
      trow = trow < (T_ - 1) ? trow : (T_ - 1);
      size_t gi = ((size_t)(n * T_ + trow) << 6) + cir;
#if HAVE_ASYNC_LDS
      __builtin_amdgcn_global_load_async_to_lds_b128(
          (as1_v4i_ptr)(gsrc + gi),
          (as3_v4i_ptr)&atile_i4[i],
          /*offset=*/0, /*cpol=*/0);
#else
      atile_i4[i] = gsrc[gi];
#endif
    }
#if HAVE_ASYNC_LDS
#if __has_builtin(__builtin_amdgcn_s_wait_asynccnt)
    __builtin_amdgcn_s_wait_asynccnt(0);
#else
    asm volatile("s_wait_asynccnt 0x0" ::: "memory");
#endif
#endif
    __syncthreads();
  }

  const uint16_t* atile = (const uint16_t*)atile_i4;

  // A fragment source (ISA 16-bit A 16x32 layout): lanes<16 hold K = k0..k0+7
  // and k0+16..k0+23 of row m=lane; lanes>=16 hold K = k0+8..15, k0+24..31.
  const int aoff = hb * 8;

  // B fragment source: column e of B == row e of W_key.
  // lanes 0-15: K=k0..k0+7, k0+16..23 of col=lane; lanes 16-31: K=k0+8..15, k0+24..31 of col=lane-16.
  const uint16_t* brow[4];
#pragma unroll
  for (int ct = 0; ct < 4; ++ct) {
    int e = h * A_ + wave * 64 + ct * 16 + lh;
    brow[ct] = wkeyb + (size_t)e * ED_;
  }
  const int boff = hb * 8;

  v8f acc[8] = {};   // [rt*4 + ct]

  for (int k0 = 0; k0 < ED_; k0 += 32) {
    // Two A fragments from LDS (rows lh and lh+16 of the 32-row tile)
    v8bf a0lo = *(const v8bf*)(atile + (size_t)lh * ED_ + k0 + aoff);
    v8bf a0hi = *(const v8bf*)(atile + (size_t)lh * ED_ + k0 + aoff + 16);
    v16bf a0 = __builtin_shufflevector(a0lo, a0hi,
        0, 1, 2, 3, 4, 5, 6, 7, 8, 9, 10, 11, 12, 13, 14, 15);
    v8bf a1lo = *(const v8bf*)(atile + (size_t)(lh + 16) * ED_ + k0 + aoff);
    v8bf a1hi = *(const v8bf*)(atile + (size_t)(lh + 16) * ED_ + k0 + aoff + 16);
    v16bf a1 = __builtin_shufflevector(a1lo, a1hi,
        0, 1, 2, 3, 4, 5, 6, 7, 8, 9, 10, 11, 12, 13, 14, 15);
#pragma unroll
    for (int ct = 0; ct < 4; ++ct) {
      v8bf blo = *(const v8bf*)(brow[ct] + k0 + boff);
      v8bf bhi = *(const v8bf*)(brow[ct] + k0 + boff + 16);
      v16bf b = __builtin_shufflevector(blo, bhi,
          0, 1, 2, 3, 4, 5, 6, 7, 8, 9, 10, 11, 12, 13, 14, 15);
      acc[ct]     = __builtin_amdgcn_wmma_f32_16x16x32_bf16(
          false, a0, false, b, (short)0, acc[ct], false, false);
      acc[4 + ct] = __builtin_amdgcn_wmma_f32_16x16x32_bf16(
          false, a1, false, b, (short)0, acc[4 + ct], false, false);
    }
  }

  // Epilogue: C layout — lanes<16: col=lane, row m=i; lanes>=16: col=lane-16, m=i+8.
  const float* dpn = dp + (size_t)n * INNER_ + h * A_;
  const float* wc  = w_conv + (size_t)h * A_;

#pragma unroll
  for (int rt = 0; rt < 2; ++rt) {
#pragma unroll
    for (int i = 0; i < 8; ++i) {
      float s = 0.f;
#pragma unroll
      for (int ct = 0; ct < 4; ++ct) {
        int a_idx = wave * 64 + ct * 16 + lh;
        float v = acc[rt * 4 + ct][i] + dpn[a_idx];
        s += wc[a_idx] * tanhf(v);
      }
      // reduce over 16 columns: xor on low 4 lane bits stays inside each half
      s += __shfl_xor(s, 1, 32);
      s += __shfl_xor(s, 2, 32);
      s += __shfl_xor(s, 4, 32);
      s += __shfl_xor(s, 8, 32);
      if (lh == 0)  // lane 0 (rows i) and lane 16 (rows i+8)
        part[wave][rt * 16 + i + hb * 8] = s;
    }
  }
  __syncthreads();
  if (tid < MT_) {
    int t = t0 + tid;
    if (t < T_) {
      float s = 0.f;
#pragma unroll
      for (int w = 0; w < 8; ++w) s += part[w][tid];
      score[((size_t)n * H_ + h) * T_ + t] = s;
    }
  }
}

// ---------------- masked softmax over T -> ali (into d_out) ----------------
__global__ __launch_bounds__(256) void softmax_kernel(
    const float* __restrict__ score, const int* __restrict__ enc_len,
    float* __restrict__ ali)
{
  const int n = blockIdx.x, h = blockIdx.y;
  const int len = enc_len[n];
  const float* s = score + ((size_t)n * H_ + h) * T_;
  float* o = ali + ((size_t)n * H_ + h) * T_;
  __shared__ float red[256];
  const int tid = threadIdx.x;

  float m = -3.402823466e38f;
  for (int t = tid; t < len; t += 256) m = fmaxf(m, s[t]);
  red[tid] = m; __syncthreads();
  for (int st = 128; st > 0; st >>= 1) {
    if (tid < st) red[tid] = fmaxf(red[tid], red[tid + st]);
    __syncthreads();
  }
  m = red[0]; __syncthreads();

  float sum = 0.f;
  for (int t = tid; t < T_; t += 256) {
    float e = (t < len) ? expf(s[t] - m) : 0.f;
    o[t] = e;
    sum += e;
  }
  red[tid] = sum; __syncthreads();
  for (int st = 128; st > 0; st >>= 1) {
    if (tid < st) red[tid] += red[tid + st];
    __syncthreads();
  }
  const float inv = 1.f / red[0];
  for (int t = tid; t < T_; t += 256) o[t] *= inv;
}

// ---------------- weighted enc (t-segmented, no atomics) -------------------
// wepart[n,h,d,seg] = sum_{t in seg} ali[n,h,t] * enc[n,t,d]
__global__ void wenc_partial_kernel(const float* __restrict__ ali,
                                    const float* __restrict__ enc,
                                    float* __restrict__ wepart)
{
  const int d   = blockIdx.x * 128 + threadIdx.x;  // grid.x = 4
  const int n   = blockIdx.y;
  const int seg = blockIdx.z;
  const int tb = seg * (T_ / TSEG_), te = tb + (T_ / TSEG_);
  const float* ap = ali + (size_t)n * H_ * T_;
  const float* ep = enc + (size_t)n * T_ * ED_ + d;
  float acc[H_] = {0.f, 0.f, 0.f, 0.f};
  for (int t = tb; t < te; ++t) {
    float ev = ep[(size_t)t * ED_];
#pragma unroll
    for (int h = 0; h < H_; ++h) acc[h] += ap[h * T_ + t] * ev;
  }
#pragma unroll
  for (int h = 0; h < H_; ++h)
    wepart[(((size_t)n * H_ + h) * ED_ + d) * TSEG_ + seg] = acc[h];
}

__global__ void wenc_reduce_kernel(const float* __restrict__ wepart,
                                   float* __restrict__ we) {
  int idx = blockIdx.x * 256 + threadIdx.x;  // N*H*ED = 32768
  float s = 0.f;
#pragma unroll
  for (int seg = 0; seg < TSEG_; ++seg) s += wepart[(size_t)idx * TSEG_ + seg];
  we[idx] = s;
}

// ---------------- ctx_pre[n,e] = we[n,h(e),:] . W_enc[e,:] + b_enc[e] ------
__global__ void ctxpre_kernel(const float* __restrict__ we,
                              const float* __restrict__ W_enc,
                              const float* __restrict__ b_enc,
                              float* __restrict__ cpre) {
  int idx = blockIdx.x * 256 + threadIdx.x;  // N*INNER
  int n = idx >> 11;
  int e = idx & (INNER_ - 1);
  int h = e >> 9;
  const float* wv = we + ((size_t)n * H_ + h) * ED_;
  const float* wr = W_enc + (size_t)e * ED_;
  float s = b_enc[e];
#pragma unroll 8
  for (int d = 0; d < ED_; ++d) s += wv[d] * wr[d];
  cpre[idx] = s;
}

// ---------------- ctx[n,dd] = cpre[n,:] . W_ctx[dd,:] + b_ctx[dd] ----------
__global__ void ctxout_kernel(const float* __restrict__ cpre,
                              const float* __restrict__ W_ctx,
                              const float* __restrict__ b_ctx,
                              float* __restrict__ out) {
  int idx = blockIdx.x * 256 + threadIdx.x;  // N*ED = 8192
  int n = idx >> 9;
  int dd = idx & (ED_ - 1);
  const float* x = cpre + (size_t)n * INNER_;
  const float* w = W_ctx + (size_t)dd * INNER_;
  float s = b_ctx[dd];
#pragma unroll 8
  for (int e = 0; e < INNER_; ++e) s += x[e] * w[e];
  out[idx] = s;
}

extern "C" void kernel_launch(void* const* d_in, const int* in_sizes, int n_in,
                              void* d_out, int out_size, void* d_ws, size_t ws_size,
                              hipStream_t stream) {
  const float* enc_pad  = (const float*)d_in[0];
  const int*   enc_len  = (const int*)d_in[1];
  const float* dec_prev = (const float*)d_in[2];
  /* d_in[3] = ali_prev, unused by the module */
  const float* W_enc  = (const float*)d_in[4];
  const float* b_enc  = (const float*)d_in[5];
  const float* W_key  = (const float*)d_in[6];
  const float* W_dec  = (const float*)d_in[7];
  const float* W_ctx  = (const float*)d_in[8];
  const float* b_ctx  = (const float*)d_in[9];
  const float* w_conv = (const float*)d_in[10];

  // Workspace layout (~37.6 MB total)
  char* ws = (char*)d_ws;
  uint16_t* encb  = (uint16_t*)ws;                              // 32,768,000 B
  uint16_t* wkeyb = (uint16_t*)(ws + 33554432);                 //  2,097,152 B
  float* dp     = (float*)(ws + 33554432 + 2097152);            // N*INNER
  float* score  = dp + (size_t)N_ * INNER_;                     // N*H*T
  float* wepart = score + (size_t)N_ * H_ * T_;                 // N*H*ED*TSEG
  float* we     = wepart + (size_t)N_ * H_ * ED_ * TSEG_;       // N*H*ED
  float* cpre   = we + (size_t)N_ * H_ * ED_;                   // N*INNER

  float* ali_out = (float*)d_out;                         // N*H*T = 128000
  float* ctx_out = (float*)d_out + (size_t)N_ * H_ * T_;  // N*ED

  const int enc_q  = N_ * T_ * ED_ / 4;   // float4 count
  const int wkey_q = INNER_ * ED_ / 4;

  cvt_bf16_kernel<<<(enc_q + 255) / 256, 256, 0, stream>>>(enc_pad, encb, enc_q);
  cvt_bf16_kernel<<<(wkey_q + 255) / 256, 256, 0, stream>>>(W_key, wkeyb, wkey_q);
  decpart_kernel<<<(N_ * INNER_) / 256, 256, 0, stream>>>(dec_prev, W_dec, dp);

  score_kernel<<<dim3((T_ + MT_ - 1) / MT_, N_, H_), 256, 0, stream>>>(
      encb, wkeyb, dp, w_conv, score);

  softmax_kernel<<<dim3(N_, H_), 256, 0, stream>>>(score, enc_len, ali_out);

  wenc_partial_kernel<<<dim3(ED_ / 128, N_, TSEG_), 128, 0, stream>>>(ali_out, enc_pad, wepart);
  wenc_reduce_kernel<<<(N_ * H_ * ED_) / 256, 256, 0, stream>>>(wepart, we);
  ctxpre_kernel<<<(N_ * INNER_) / 256, 256, 0, stream>>>(we, W_enc, b_enc, cpre);
  ctxout_kernel<<<(N_ * ED_) / 256, 256, 0, stream>>>(cpre, W_ctx, b_ctx, ctx_out);
}
